// MultiHeadAttention_11957188952536
// MI455X (gfx1250) — matrix operand order repaired and checked by reading, not compiled, not run on hardware
//
#include <hip/hip_runtime.h>
#include <hip/hip_bf16.h>

// ---------------------------------------------------------------------------
// MHA block for MI455X (gfx1250): bf16 WMMA everywhere, flash-attention inner
// loop with double-buffered async-to-LDS K/V staging, fused Wo + residual +
// LayerNorm epilogue.
// ---------------------------------------------------------------------------

typedef __bf16 bf16_t;
typedef __attribute__((ext_vector_type(8)))  bf16_t v8bf;
typedef __attribute__((ext_vector_type(16))) bf16_t v16bf;
typedef __attribute__((ext_vector_type(8)))  float  v8f;
typedef __attribute__((ext_vector_type(4)))  int    v4i_t;

#define N_TOK 4096
#define D_IN  1024
#define DH    128
#define NH    8
#define INV_T (1.0f / 32.0f)   // T = sqrt(1024)

#if !__has_builtin(__builtin_amdgcn_global_load_async_to_lds_b128)
#warning "gfx1250 async-to-LDS builtin unavailable; using synchronous fallback"
#endif

static __device__ __forceinline__ v16bf cat16(v8bf lo, v8bf hi) {
  return __builtin_shufflevector(lo, hi, 0,1,2,3,4,5,6,7,8,9,10,11,12,13,14,15);
}

// A-fragment (16 rows x 32 K, bf16) from row-major storage, leading dim ld.
// Lane l holds row l%16; K elements base..base+7 and base+16..base+23,
// base = (l/16)*8  (ISA 16-bit A 16x32 layout).
static __device__ __forceinline__ v16bf load_a(const bf16_t* __restrict__ src,
                                               int ld, int lane) {
  const bf16_t* p = src + (lane & 15) * ld + ((lane >> 4) << 3);
  v8bf lo = *(const v8bf*)p;
  v8bf hi = *(const v8bf*)(p + 16);
  return cat16(lo, hi);
}

// B-fragment (32 K x 16 cols, bf16) from B^T storage (col-major B == row-major
// B^T with leading dim ld). Lane l holds col l%16; K elements base..base+7 and
// base+16..base+23, base = (l/16)*8 (same K order as load_a).
static __device__ __forceinline__ v16bf load_bt(const bf16_t* __restrict__ srcT,
                                                int ld, int lane) {
  const bf16_t* p = srcT + (lane & 15) * ld + ((lane >> 4) << 3);
  v8bf lo = *(const v8bf*)p;
  v8bf hi = *(const v8bf*)(p + 16);
  return cat16(lo, hi);
}

static __device__ __forceinline__ v8f wmma_bf16(v16bf a, v16bf b, v8f c) {
  return __builtin_amdgcn_wmma_f32_16x16x32_bf16(false, a, false, b,
                                                 (short)0, c, false, false);
}

// 16-byte async global->LDS copy (ASYNCcnt-tracked on gfx1250).
// Builtin prototype (from clang diagnostics): (v4i AS1*, v4i AS3*, imm, imm).
static __device__ __forceinline__ void async_copy16(const bf16_t* gsrc,
                                                    bf16_t* ldst) {
#if __has_builtin(__builtin_amdgcn_global_load_async_to_lds_b128)
  __builtin_amdgcn_global_load_async_to_lds_b128(
      (__attribute__((address_space(1))) v4i_t*)gsrc,
      (__attribute__((address_space(3))) v4i_t*)ldst, 0, 0);
#else
  *(v8bf*)ldst = *(const v8bf*)gsrc;
#endif
}

static __device__ __forceinline__ void wait_async0() {
#if __has_builtin(__builtin_amdgcn_s_wait_asynccnt)
  __builtin_amdgcn_s_wait_asynccnt(0);
#elif __has_builtin(__builtin_amdgcn_global_load_async_to_lds_b128)
  asm volatile("s_wait_asynccnt 0" ::: "memory");
#endif
}

// ---------------------------------------------------------------------------
// Kernel 1a: fp32 -> bf16 pack of x
__global__ void __launch_bounds__(256) pack_x_kernel(const float* __restrict__ x,
                                                     bf16_t* __restrict__ xb) {
  int i = blockIdx.x * 256 + threadIdx.x;
  xb[i] = (bf16_t)x[i];
}

// Kernel 1b: fp32 [in][out] weight -> bf16 transposed [out][in]
__global__ void __launch_bounds__(256) transpose_w_kernel(const float* __restrict__ w,
                                                          bf16_t* __restrict__ wT) {
  int idx = blockIdx.x * 256 + threadIdx.x;   // idx = o*1024 + i
  int o = idx >> 10, i = idx & 1023;
  wT[idx] = (bf16_t)w[i * D_IN + o];
}

// ---------------------------------------------------------------------------
// Kernel 2: fused QKV projection.  Output cols 0..3071 = [Q|K|V].
// Scatter-stores into head-major layouts:
//   Q: [h][tok][d]   K: [h][tok][d] (== B^T for Q·Kᵀ)   V: [h][d][tok] (== B^T for P·V)
// Block = 256 thr (8 waves); block tile 64(M) x 64(N); wave tile 16 x 32.
__global__ void __launch_bounds__(256) qkv_gemm_kernel(
    const bf16_t* __restrict__ xb,  const bf16_t* __restrict__ WqT,
    const bf16_t* __restrict__ WkT, const bf16_t* __restrict__ WvT,
    bf16_t* __restrict__ Qb, bf16_t* __restrict__ Kb, bf16_t* __restrict__ Vt) {
  int lane = threadIdx.x & 31;
  int w    = threadIdx.x >> 5;
  int wm = w >> 1, wn = w & 1;
  int m0 = blockIdx.x * 64 + wm * 16;
  int ncol0 = blockIdx.y * 64 + wn * 32;      // 0..3071
  int mat = ncol0 >> 10;                       // 0=Q 1=K 2=V
  int cl0 = ncol0 & 1023;
  const bf16_t* WT = (mat == 0) ? WqT : (mat == 1) ? WkT : WvT;

  v8f acc0 = {}; v8f acc1 = {};
  for (int kk = 0; kk < D_IN; kk += 32) {
    __builtin_prefetch(xb + m0 * D_IN + kk + 64, 0, 3);
    v16bf a  = load_a (xb + m0 * D_IN + kk, D_IN, lane);
    v16bf b0 = load_bt(WT + (cl0 +  0) * D_IN + kk, D_IN, lane);
    v16bf b1 = load_bt(WT + (cl0 + 16) * D_IN + kk, D_IN, lane);
    acc0 = wmma_bf16(a, b0, acc0);
    acc1 = wmma_bf16(a, b1, acc1);
  }
  int hi8 = (lane >> 4) << 3;
#pragma unroll
  for (int v = 0; v < 8; ++v) {
    int m = m0 + v + hi8;
    int c0 = cl0 + (lane & 15);
    int c1 = c0 + 16;
    int d0 = c0 >> 3, h0 = c0 & 7;
    int d1 = c1 >> 3, h1 = c1 & 7;
    bf16_t v0 = (bf16_t)acc0[v];
    bf16_t v1 = (bf16_t)acc1[v];
    if (mat == 0) {
      Qb[(h0 * N_TOK + m) * DH + d0] = v0;
      Qb[(h1 * N_TOK + m) * DH + d1] = v1;
    } else if (mat == 1) {
      Kb[(h0 * N_TOK + m) * DH + d0] = v0;
      Kb[(h1 * N_TOK + m) * DH + d1] = v1;
    } else {
      Vt[(h0 * DH + d0) * N_TOK + m] = v0;
      Vt[(h1 * DH + d1) * N_TOK + m] = v1;
    }
  }
}

// ---------------------------------------------------------------------------
// Kernel 3: flash attention. Block = 4 waves = 64 queries of one head.
// K/V tiles (32 keys) staged into LDS via double-buffered async copies so all
// 4 waves share one global fetch; WMMA B-fragments come from ds_load_b128.
__global__ void __launch_bounds__(128) attention_kernel(
    const bf16_t* __restrict__ Qb, const bf16_t* __restrict__ Kb,
    const bf16_t* __restrict__ Vt, bf16_t* __restrict__ attn) {
  __shared__ bf16_t kt[2][32 * 128];      // [buf][j_local][d]   8KB each
  __shared__ bf16_t vt[2][128 * 32];      // [buf][d][j_local]   8KB each
  __shared__ bf16_t plds[4][16 * 32];     // per-wave P bounce (C->A layout)
  int tid  = threadIdx.x;
  int lane = tid & 31;
  int w    = tid >> 5;
  int h  = blockIdx.x >> 6;
  int qb = blockIdx.x & 63;
  int m0 = qb * 64 + w * 16;
  int hi8 = (lane >> 4) << 3;

  // stage one 32-key K tile (contiguous 8KB) + V^T tile (128 rows x 64B)
  auto stage = [&](int j0, int buf) {
    const bf16_t* ks = Kb + (h * N_TOK + j0) * DH;   // 32*128 contiguous halves
#pragma unroll
    for (int i = 0; i < 4; ++i) {
      int t = i * 128 + tid;                         // 0..511 chunks of 8 halves
      async_copy16(ks + t * 8, &kt[buf][t * 8]);
    }
#pragma unroll
    for (int i = 0; i < 4; ++i) {
      int t = i * 128 + tid;
      int d = t >> 2, jc = (t & 3) * 8;
      async_copy16(Vt + (h * DH + d) * N_TOK + j0 + jc, &vt[buf][d * 32 + jc]);
    }
  };

  // Q fragments for 16 rows x 128 d (4 A-frags, resident in VGPRs)
  v16bf qf[4];
#pragma unroll
  for (int dk = 0; dk < 4; ++dk)
    qf[dk] = load_a(Qb + (h * N_TOK + m0) * DH + dk * 32, DH, lane);

  v8f o_acc[8];
#pragma unroll
  for (int dt = 0; dt < 8; ++dt) o_acc[dt] = (v8f){};
  float m_r[8], l_r[8];
#pragma unroll
  for (int v = 0; v < 8; ++v) { m_r[v] = -INFINITY; l_r[v] = 0.0f; }

  stage(0, 0);                         // prologue copy for tile 0

  for (int j0 = 0; j0 < N_TOK; j0 += 32) {
    int buf = (j0 >> 5) & 1;
    wait_async0();                     // this wave's copy of tile j0 done
    __syncthreads();                   // all waves' copies done; prev readers done
    if (j0 + 32 < N_TOK) stage(j0 + 32, buf ^ 1);   // overlap next copy w/ compute

    // ---- S = Q K^T for 16 x 32 keys (fp32 accum), B-frags from LDS
    v8f s0 = {}; v8f s1 = {};
#pragma unroll
    for (int dk = 0; dk < 4; ++dk) {
      v16bf bk0 = load_bt(&kt[buf][ 0 * 128] + dk * 32, 128, lane);
      v16bf bk1 = load_bt(&kt[buf][16 * 128] + dk * 32, 128, lane);
      s0 = wmma_bf16(qf[dk], bk0, s0);
      s1 = wmma_bf16(qf[dk], bk1, s1);
    }
    // ---- online softmax (per-row reductions across 16-lane halves)
    float corr[8];
#pragma unroll
    for (int v = 0; v < 8; ++v) {
      float a0 = s0[v] * INV_T;
      float a1 = s1[v] * INV_T;
      float mx = fmaxf(a0, a1);
      mx = fmaxf(mx, __shfl_xor(mx, 1, 32));
      mx = fmaxf(mx, __shfl_xor(mx, 2, 32));
      mx = fmaxf(mx, __shfl_xor(mx, 4, 32));
      mx = fmaxf(mx, __shfl_xor(mx, 8, 32));
      float mn = fmaxf(m_r[v], mx);
      corr[v] = __expf(m_r[v] - mn);
      m_r[v] = mn;
      float p0 = __expf(a0 - mn);
      float p1 = __expf(a1 - mn);
      float rs = p0 + p1;
      rs += __shfl_xor(rs, 1, 32);
      rs += __shfl_xor(rs, 2, 32);
      rs += __shfl_xor(rs, 4, 32);
      rs += __shfl_xor(rs, 8, 32);
      l_r[v] = l_r[v] * corr[v] + rs;
      // C-layout -> LDS (row-major 16x32 P tile), wave-private region
      plds[w][(v + hi8) * 32 +      (lane & 15)] = (bf16_t)p0;
      plds[w][(v + hi8) * 32 + 16 + (lane & 15)] = (bf16_t)p1;
    }
    // rescale running O by correction factor
#pragma unroll
    for (int dt = 0; dt < 8; ++dt)
#pragma unroll
      for (int v = 0; v < 8; ++v) o_acc[dt][v] *= corr[v];
    // P back as A-fragment (same-wave LDS dependency -> dscnt wait)
    v16bf pf = load_a(&plds[w][0], 32, lane);
    // ---- O += P · V, B-frags from LDS V^T tile
#pragma unroll
    for (int dt = 0; dt < 8; ++dt) {
      v16bf bv = load_bt(&vt[buf][(dt * 16) * 32], 32, lane);
      o_acc[dt] = wmma_bf16(pf, bv, o_acc[dt]);
    }
  }

  // ---- normalize and scatter to interleaved [i][d*8+h]
  float invl[8];
#pragma unroll
  for (int v = 0; v < 8; ++v) invl[v] = 1.0f / l_r[v];
#pragma unroll
  for (int dt = 0; dt < 8; ++dt) {
    int d = dt * 16 + (lane & 15);
#pragma unroll
    for (int v = 0; v < 8; ++v) {
      int i = m0 + v + hi8;
      attn[i * D_IN + d * 8 + h] = (bf16_t)(o_acc[dt][v] * invl[v]);
    }
  }
}

// ---------------------------------------------------------------------------
// Kernel 4: out = attn @ Wo + residual, then LayerNorm over 1024 cols.
// Block = 256 thr (8 waves) owns 16 full rows; wave w covers cols w*128..+127.
__global__ void __launch_bounds__(256) out_ln_kernel(
    const bf16_t* __restrict__ attn, const bf16_t* __restrict__ WoT,
    const float* __restrict__ x, const float* __restrict__ gamma,
    const float* __restrict__ beta, float* __restrict__ out) {
  __shared__ float ssum[8][16];
  __shared__ float ssq[8][16];
  int lane = threadIdx.x & 31;
  int w    = threadIdx.x >> 5;
  int m0 = blockIdx.x * 16;
  int n0 = w * 128;
  int hi8 = (lane >> 4) << 3;

  v8f acc[8];
#pragma unroll
  for (int dt = 0; dt < 8; ++dt) acc[dt] = (v8f){};
  for (int kk = 0; kk < D_IN; kk += 32) {
    __builtin_prefetch(attn + m0 * D_IN + kk + 64, 0, 3);
    v16bf a = load_a(attn + m0 * D_IN + kk, D_IN, lane);
#pragma unroll
    for (int dt = 0; dt < 8; ++dt) {
      v16bf b = load_bt(WoT + (n0 + dt * 16) * D_IN + kk, D_IN, lane);
      acc[dt] = wmma_bf16(a, b, acc[dt]);
    }
  }

  // residual + per-row partial stats over this wave's 128 cols
  float psum[8], psq[8];
#pragma unroll
  for (int v = 0; v < 8; ++v) { psum[v] = 0.0f; psq[v] = 0.0f; }
#pragma unroll
  for (int dt = 0; dt < 8; ++dt) {
    int col = n0 + dt * 16 + (lane & 15);
#pragma unroll
    for (int v = 0; v < 8; ++v) {
      int m = m0 + v + hi8;
      float val = acc[dt][v] + x[m * D_IN + col];
      acc[dt][v] = val;
      psum[v] += val;
      psq[v]  += val * val;
    }
  }
#pragma unroll
  for (int v = 0; v < 8; ++v) {
    psum[v] += __shfl_xor(psum[v], 1, 32);  psq[v] += __shfl_xor(psq[v], 1, 32);
    psum[v] += __shfl_xor(psum[v], 2, 32);  psq[v] += __shfl_xor(psq[v], 2, 32);
    psum[v] += __shfl_xor(psum[v], 4, 32);  psq[v] += __shfl_xor(psq[v], 4, 32);
    psum[v] += __shfl_xor(psum[v], 8, 32);  psq[v] += __shfl_xor(psq[v], 8, 32);
    if ((lane & 15) == 0) {
      ssum[w][v + hi8] = psum[v];
      ssq [w][v + hi8] = psq[v];
    }
  }
  __syncthreads();
  float mu[8], rsg[8];
#pragma unroll
  for (int v = 0; v < 8; ++v) {
    int rl = v + hi8;
    float s = 0.0f, s2 = 0.0f;
#pragma unroll
    for (int ww = 0; ww < 8; ++ww) { s += ssum[ww][rl]; s2 += ssq[ww][rl]; }
    float m = s * (1.0f / D_IN);
    float var = s2 * (1.0f / D_IN) - m * m;
    mu[v] = m;
    rsg[v] = rsqrtf(var + 1e-5f);
  }
#pragma unroll
  for (int dt = 0; dt < 8; ++dt) {
    int col = n0 + dt * 16 + (lane & 15);
    float g = gamma[col], bt = beta[col];
#pragma unroll
    for (int v = 0; v < 8; ++v) {
      int m = m0 + v + hi8;
      out[m * D_IN + col] = (acc[dt][v] - mu[v]) * rsg[v] * g + bt;
    }
  }
}

// ---------------------------------------------------------------------------
extern "C" void kernel_launch(void* const* d_in, const int* in_sizes, int n_in,
                              void* d_out, int out_size, void* d_ws, size_t ws_size,
                              hipStream_t stream) {
  const float* x     = (const float*)d_in[0];
  // d_in[1] = edge_attr (unused by reference)
  const float* W_q   = (const float*)d_in[2];
  const float* W_k   = (const float*)d_in[3];
  const float* W_v   = (const float*)d_in[4];
  const float* W_o   = (const float*)d_in[5];
  const float* gamma = (const float*)d_in[6];
  const float* beta  = (const float*)d_in[7];
  float* out = (float*)d_out;

  char* ws = (char*)d_ws;
  const size_t MB = 1024 * 1024;
  bf16_t* xb   = (bf16_t*)(ws + 0 * MB);    // 8 MiB  [4096][1024]
  bf16_t* WqT  = (bf16_t*)(ws + 8 * MB);    // 2 MiB  [1024][1024]
  bf16_t* WkT  = (bf16_t*)(ws + 10 * MB);
  bf16_t* WvT  = (bf16_t*)(ws + 12 * MB);
  bf16_t* WoT  = (bf16_t*)(ws + 14 * MB);
  bf16_t* Qb   = (bf16_t*)(ws + 16 * MB);   // 8 MiB  [8][4096][128]
  bf16_t* Kb   = (bf16_t*)(ws + 24 * MB);   // 8 MiB  [8][4096][128]
  bf16_t* Vt   = (bf16_t*)(ws + 32 * MB);   // 8 MiB  [8][128][4096]
  bf16_t* attn = (bf16_t*)(ws + 40 * MB);   // 8 MiB  [4096][1024]

  pack_x_kernel<<<(N_TOK * D_IN) / 256, 256, 0, stream>>>(x, xb);
  transpose_w_kernel<<<(D_IN * D_IN) / 256, 256, 0, stream>>>(W_q, WqT);
  transpose_w_kernel<<<(D_IN * D_IN) / 256, 256, 0, stream>>>(W_k, WkT);
  transpose_w_kernel<<<(D_IN * D_IN) / 256, 256, 0, stream>>>(W_v, WvT);
  transpose_w_kernel<<<(D_IN * D_IN) / 256, 256, 0, stream>>>(W_o, WoT);

  qkv_gemm_kernel<<<dim3(N_TOK / 64, (3 * D_IN) / 64), 256, 0, stream>>>(
      xb, WqT, WkT, WvT, Qb, Kb, Vt);

  attention_kernel<<<NH * (N_TOK / 64), 128, 0, stream>>>(Qb, Kb, Vt, attn);

  out_ln_kernel<<<N_TOK / 16, 256, 0, stream>>>(attn, WoT, x, gamma, beta, out);
}
